// VGAEModel_54142357733691
// MI455X (gfx1250) — matrix operand order repaired and checked by reading, not compiled, not run on hardware
//
#include <hip/hip_runtime.h>
#include <hip/hip_bf16.h>

typedef __attribute__((ext_vector_type(16))) __bf16 v16bf;
typedef __attribute__((ext_vector_type(8)))  float  v8f;

__device__ __forceinline__ __bf16 f2bf(float f) {
    unsigned u = __builtin_bit_cast(unsigned, f);
    u += 0x7FFFu + ((u >> 16) & 1u);            // round-to-nearest-even
    unsigned short s = (unsigned short)(u >> 16);
    return __builtin_bit_cast(__bf16, s);
}
__device__ __forceinline__ float bf2f(__bf16 b) {
    unsigned short s = __builtin_bit_cast(unsigned short, b);
    unsigned u = ((unsigned)s) << 16;
    return __builtin_bit_cast(float, u);
}

// ---------------- utility kernels ----------------

__global__ void zero_f32(float* __restrict__ p, long count) {
    long i = (long)blockIdx.x * blockDim.x + threadIdx.x;
    long stride = (long)gridDim.x * blockDim.x;
    for (; i < count; i += stride) p[i] = 0.0f;
}

__global__ void deg_accum(const int* __restrict__ src, const int* __restrict__ dst,
                          float* __restrict__ deg_out, float* __restrict__ deg_in, int E) {
    int e = blockIdx.x * blockDim.x + threadIdx.x;
    if (e < E) {
        atomicAdd(&deg_out[src[e]], 1.0f);
        atomicAdd(&deg_in[dst[e]], 1.0f);
    }
}

__global__ void deg_finalize(float* __restrict__ d, long count) {
    long i = (long)blockIdx.x * blockDim.x + threadIdx.x;
    if (i < count) d[i] = rsqrtf(fmaxf(d[i], 1.0f));
}

// Transpose weights to [n][k] and split fp32 -> bf16 hi/lo pairs.
// W1 [512,64] -> w1t_* [64][512]; W23 = [W2 | W3] [64,64] -> w23t_* [64][64]
__global__ void prep_weights(const float* __restrict__ W1,
                             const float* __restrict__ W2,
                             const float* __restrict__ W3,
                             __bf16* __restrict__ w1t_hi, __bf16* __restrict__ w1t_lo,
                             __bf16* __restrict__ w23t_hi, __bf16* __restrict__ w23t_lo) {
    int t = blockIdx.x * blockDim.x + threadIdx.x;
    if (t < 512 * 64) {
        int k = t >> 6, n = t & 63;
        float v = W1[t];                       // W1[k*64+n]
        __bf16 hi = f2bf(v);
        __bf16 lo = f2bf(v - bf2f(hi));
        w1t_hi[n * 512 + k] = hi;
        w1t_lo[n * 512 + k] = lo;
    } else if (t < 512 * 64 + 64 * 64) {
        int t2 = t - 512 * 64;
        int k = t2 >> 6, n = t2 & 63;
        float v = (n < 32) ? W2[k * 32 + n] : W3[k * 32 + (n - 32)];
        __bf16 hi = f2bf(v);
        __bf16 lo = f2bf(v - bf2f(hi));
        w23t_hi[n * 64 + k] = hi;
        w23t_lo[n * 64 + k] = lo;
    }
}

// ---------------- WMMA GEMM: Out[r, 0..63] = (A[r,:] * scale[r]) @ B ----------------
// bf16x3 emulation for ~fp32 accuracy. One wave -> 16 rows x 64 cols (4 accum tiles).
// A-fragment layout (16-bit A 16x32, contiguous k order, same as B): lane l: m=l&15, h=l>>4,
//   slots 0..7  <- K = kb + h*16 + j
//   slots 8..15 <- K = kb + h*16 + 8 + j
// B-fragment (32x16): lane col n = l&15, slot j <- K = kb + 16*h + j  (contiguous in B^T)
// C/D (f32 16x16): slot i -> row M = 8*h + i, col N = l&15.
// WMMAs are issued term-major across the 4 independent accumulators so that
// dependent WMMAs on the same accumulator are >=4 instructions apart,
// covering the bf16 WMMA->WMMA RAW hazard window with useful work (no v_nop).
__global__ __launch_bounds__(128)
void gemm_bf16x3(const float* __restrict__ A, const float* __restrict__ scale,
                 const __bf16* __restrict__ Bhi, const __bf16* __restrict__ Blo,
                 float* __restrict__ Out, int nrows, int K) {
    const int wave = threadIdx.x >> 5;
    const int lane = threadIdx.x & 31;
    const int m = lane & 15;
    const int h = lane >> 4;
    const int row_base = blockIdx.x * 64 + wave * 16;

    int row = row_base + m;
    if (row >= nrows) row = nrows - 1;            // clamp: keep EXEC all-ones for WMMA
    const float sc = scale[row];
    const float* __restrict__ arow = A + (size_t)row * K;

    v8f acc[4] = {};

    for (int kb = 0; kb < K; kb += 32) {
        // ---- A fragment (hi/lo split) ----
        float va[16];
        {
            const float4* p0 = (const float4*)(arow + kb + h * 16);
            const float4* p1 = (const float4*)(arow + kb + h * 16 + 8);
            float4 x;
            x = p0[0]; va[0] = x.x; va[1]  = x.y; va[2]  = x.z; va[3]  = x.w;
            x = p0[1]; va[4] = x.x; va[5]  = x.y; va[6]  = x.z; va[7]  = x.w;
            x = p1[0]; va[8] = x.x; va[9]  = x.y; va[10] = x.z; va[11] = x.w;
            x = p1[1]; va[12] = x.x; va[13] = x.y; va[14] = x.z; va[15] = x.w;
        }
        v16bf a_hi, a_lo;
        #pragma unroll
        for (int j = 0; j < 16; ++j) {
            float f = va[j] * sc;
            __bf16 hi = f2bf(f);
            a_hi[j] = hi;
            a_lo[j] = f2bf(f - bf2f(hi));
        }

        // ---- load all B fragments for this K-step ----
        v16bf bh[4], bl[4];
        #pragma unroll
        for (int t = 0; t < 4; ++t) {
            const size_t boff = (size_t)(t * 16 + m) * K + kb + h * 16;
            bh[t] = *(const v16bf*)(Bhi + boff);
            bl[t] = *(const v16bf*)(Blo + boff);
        }

        // ---- 12 WMMAs, term-major: 4 independent accs between dependent ops ----
        #pragma unroll
        for (int t = 0; t < 4; ++t)
            acc[t] = __builtin_amdgcn_wmma_f32_16x16x32_bf16(
                false, a_hi, false, bh[t], (short)0, acc[t], false, false);
        #pragma unroll
        for (int t = 0; t < 4; ++t)
            acc[t] = __builtin_amdgcn_wmma_f32_16x16x32_bf16(
                false, a_lo, false, bh[t], (short)0, acc[t], false, false);
        #pragma unroll
        for (int t = 0; t < 4; ++t)
            acc[t] = __builtin_amdgcn_wmma_f32_16x16x32_bf16(
                false, a_hi, false, bl[t], (short)0, acc[t], false, false);
    }

    // ---- store 16x64 tile ----
    #pragma unroll
    for (int t = 0; t < 4; ++t) {
        #pragma unroll
        for (int i = 0; i < 8; ++i) {
            int r = row_base + h * 8 + i;
            if (r < nrows) Out[(size_t)r * 64 + t * 16 + m] = acc[t][i];
        }
    }
}

// ---------------- edge scatter: Agg[dst] += P[src] (64 channels/edge) ----------------
__global__ void spmm_scatter(const float* __restrict__ P, const int* __restrict__ src,
                             const int* __restrict__ dst, float* __restrict__ Agg, int E) {
    long t = (long)blockIdx.x * blockDim.x + threadIdx.x;
    int e = (int)(t >> 4);
    if (e >= E) return;
    int c4 = ((int)t & 15) * 4;
    int s = src[e], d = dst[e];
    float4 v = *(const float4*)(P + (size_t)s * 64 + c4);
    float* o = Agg + (size_t)d * 64 + c4;
    atomicAdd(o + 0, v.x);
    atomicAdd(o + 1, v.y);
    atomicAdd(o + 2, v.z);
    atomicAdd(o + 3, v.w);
}

// ---------------- h = relu(agg * in_isqrt + b1) ----------------
__global__ void act_relu(const float* __restrict__ Agg, const float* __restrict__ isq_in,
                         const float* __restrict__ b1, float* __restrict__ H, long n) {
    long t = (long)blockIdx.x * blockDim.x + threadIdx.x;
    if (t >= n * 64) return;
    long i = t >> 6;
    int c = (int)(t & 63);
    float v = Agg[t] * isq_in[i] + b1[c];
    H[t] = fmaxf(v, 0.0f);
}

// ---------------- final: mean/log_std/z ----------------
__global__ void vgae_final(const float* __restrict__ Agg2, const float* __restrict__ isq_in,
                           const float* __restrict__ b2, const float* __restrict__ b3,
                           const float* __restrict__ noise, float* __restrict__ out, long n) {
    long t = (long)blockIdx.x * blockDim.x + threadIdx.x;
    if (t >= n * 32) return;
    long i = t >> 5;
    int c = (int)(t & 31);
    float s = isq_in[i];
    float mean = Agg2[i * 64 + c] * s + b2[c];
    float ls   = Agg2[i * 64 + 32 + c] * s + b3[c];
    float z = mean + noise[t] * expf(ls);
    out[t] = z;                         // z
    out[n * 32 + t] = mean;             // mean
    out[2 * n * 32 + t] = ls;           // log_std
}

extern "C" void kernel_launch(void* const* d_in, const int* in_sizes, int n_in,
                              void* d_out, int out_size, void* d_ws, size_t ws_size,
                              hipStream_t stream) {
    const float* features = (const float*)d_in[0];
    const int*   src      = (const int*)d_in[1];
    const int*   dst      = (const int*)d_in[2];
    const float* noise    = (const float*)d_in[3];
    const float* W1 = (const float*)d_in[4];
    const float* b1 = (const float*)d_in[5];
    const float* W2 = (const float*)d_in[6];
    const float* b2 = (const float*)d_in[7];
    const float* W3 = (const float*)d_in[8];
    const float* b3 = (const float*)d_in[9];

    const long n = in_sizes[0] / 512;     // 100000 nodes
    const int  E = in_sizes[1];           // 800000 edges

    // workspace layout
    float* wsf      = (float*)d_ws;
    float* isq_out  = wsf;                 // n  (deg_out -> out_isqrt, in place)
    float* isq_in   = wsf + n;             // n
    float* bufA     = wsf + 2 * n;         // n*64 : h_pre -> h -> agg2
    float* bufB     = bufA + 64 * n;       // n*64 : agg1 -> p
    __bf16* w1t_hi  = (__bf16*)(bufB + 64 * n);
    __bf16* w1t_lo  = w1t_hi + 64 * 512;
    __bf16* w23t_hi = w1t_lo + 64 * 512;
    __bf16* w23t_lo = w23t_hi + 64 * 64;

    const int gemm_blocks = (int)((n + 63) / 64);

    // degrees -> isqrt norms
    zero_f32<<<(unsigned)((2 * n + 255) / 256), 256, 0, stream>>>(isq_out, 2 * n);
    deg_accum<<<(E + 255) / 256, 256, 0, stream>>>(src, dst, isq_out, isq_in, E);
    deg_finalize<<<(unsigned)((2 * n + 255) / 256), 256, 0, stream>>>(isq_out, 2 * n);

    // weight transpose + bf16 hi/lo split
    prep_weights<<<(512 * 64 + 64 * 64 + 255) / 256, 256, 0, stream>>>(
        W1, W2, W3, w1t_hi, w1t_lo, w23t_hi, w23t_lo);

    // layer 1: h_pre = (X * out_isqrt) @ W1
    gemm_bf16x3<<<gemm_blocks, 128, 0, stream>>>(features, isq_out, w1t_hi, w1t_lo,
                                                 bufA, (int)n, 512);
    // agg1 = scatter(h_pre[src] -> dst)
    zero_f32<<<(unsigned)((64 * n + 255) / 256), 256, 0, stream>>>(bufB, 64 * n);
    spmm_scatter<<<(unsigned)(((long)E * 16 + 255) / 256), 256, 0, stream>>>(bufA, src, dst, bufB, E);
    // h = relu(agg1 * in_isqrt + b1)
    act_relu<<<(unsigned)((64 * n + 255) / 256), 256, 0, stream>>>(bufB, isq_in, b1, bufA, n);

    // layer 2+3 fused: p = (h * out_isqrt) @ [W2 | W3]
    gemm_bf16x3<<<gemm_blocks, 128, 0, stream>>>(bufA, isq_out, w23t_hi, w23t_lo,
                                                 bufB, (int)n, 64);
    // agg2 = scatter(p[src] -> dst)
    zero_f32<<<(unsigned)((64 * n + 255) / 256), 256, 0, stream>>>(bufA, 64 * n);
    spmm_scatter<<<(unsigned)(((long)E * 16 + 255) / 256), 256, 0, stream>>>(bufB, src, dst, bufA, E);

    // z / mean / log_std
    vgae_final<<<(unsigned)((32 * n + 255) / 256), 256, 0, stream>>>(
        bufA, isq_in, b2, b3, noise, (float*)d_out, n);
}
